// SparseAttention_40114994544816
// MI455X (gfx1250) — matrix operand order repaired and checked by reading, not compiled, not run on hardware
//
#include <hip/hip_runtime.h>
#include <hip/hip_bf16.h>

#define BB 4
#define SS 2048
#define EE 1024
#define TOPK 32
#define MTOT (BB * SS)   // 8192 total rows

typedef __attribute__((ext_vector_type(16))) _Float16 v16h;
typedef __attribute__((ext_vector_type(8)))  _Float16 v8h;
typedef __attribute__((ext_vector_type(8)))  float    v8f;
typedef __attribute__((ext_vector_type(4)))  float    v4f;
typedef __attribute__((ext_vector_type(4)))  int      v4i;

#if __has_builtin(__builtin_amdgcn_global_load_async_to_lds_b128) && \
    __has_builtin(__builtin_amdgcn_s_wait_asynccnt)
#define HAVE_ASYNC_LDS 1
typedef v4i __attribute__((address_space(1))) * gv4i_p;   // global int4*
typedef v4i __attribute__((address_space(3))) * lv4i_p;   // LDS int4*
#endif

// ---------------------------------------------------------------------------
// Kernel 1: QKV projection.  q[m][f] = sum_e x[m][e] * W[f][e] + b[f]
// One wave per 16x16 output tile, 8 waves -> 16x128 tile per block,
// grid.z selects {Wq,Wk,Wv}.  f16 WMMA, f32 accumulate, bias in f32.
// ---------------------------------------------------------------------------
__global__ __launch_bounds__(256) void sparse_attn_qkv(
    const float* __restrict__ x,
    const float* __restrict__ Wq, const float* __restrict__ bq,
    const float* __restrict__ Wk, const float* __restrict__ bk,
    const float* __restrict__ Wv, const float* __restrict__ bv,
    _Float16* __restrict__ Qh, _Float16* __restrict__ Kh,
    _Float16* __restrict__ Vh)
{
    const int tid  = threadIdx.x;
    const int wave = tid >> 5;
    const int lane = tid & 31;
    const int lm   = lane & 15;
    const int hi   = lane >> 4;          // 0: lanes 0-15, 1: lanes 16-31

    const int row0 = blockIdx.x * 16;
    const int f0   = blockIdx.y * 128 + wave * 16;
    const int proj = blockIdx.z;

    const float* W    = (proj == 0) ? Wq : (proj == 1) ? Wk : Wv;
    const float* bias = (proj == 0) ? bq : (proj == 1) ? bk : bv;
    _Float16*    outp = (proj == 0) ? Qh : (proj == 1) ? Kh : Vh;

    // A layout: lane holds M=lm; K runs {base..base+7, base+16..base+23}, base=8*hi
    const float* arow = x + (size_t)(row0 + lm) * EE + hi * 8;
    // B layout: lane holds N=lm; K runs {base..base+7, base+16..base+23}, base=8*hi
    const float* brow = W + (size_t)(f0 + lm) * EE + hi * 8;

    v8f c = {};
    for (int k0 = 0; k0 < EE; k0 += 32) {
        v4f a0 = *(const v4f*)(arow + k0);
        v4f a1 = *(const v4f*)(arow + k0 + 4);
        v4f a2 = *(const v4f*)(arow + k0 + 16);
        v4f a3 = *(const v4f*)(arow + k0 + 20);
        v16h a;
#pragma unroll
        for (int i = 0; i < 4; ++i) {
            a[i]      = (_Float16)a0[i];
            a[4 + i]  = (_Float16)a1[i];
            a[8 + i]  = (_Float16)a2[i];
            a[12 + i] = (_Float16)a3[i];
        }
        v4f b0 = *(const v4f*)(brow + k0);
        v4f b1 = *(const v4f*)(brow + k0 + 4);
        v4f b2 = *(const v4f*)(brow + k0 + 16);
        v4f b3 = *(const v4f*)(brow + k0 + 20);
        v16h bf;
#pragma unroll
        for (int i = 0; i < 4; ++i) {
            bf[i]      = (_Float16)b0[i];
            bf[4 + i]  = (_Float16)b1[i];
            bf[8 + i]  = (_Float16)b2[i];
            bf[12 + i] = (_Float16)b3[i];
        }
        c = __builtin_amdgcn_wmma_f32_16x16x32_f16(false, a, false, bf,
                                                   (short)0, c, false, false);
    }

    const float bval = bias[f0 + lm];
    // D layout: lane holds N=lm, VGPR r -> M = r + 8*hi
#pragma unroll
    for (int r = 0; r < 8; ++r) {
        const int m = row0 + hi * 8 + r;
        outp[(size_t)m * EE + (f0 + lm)] = (_Float16)(c[r] + bval);
    }
}

// ---------------------------------------------------------------------------
// Kernel 2: per (batch, 16-query-row strip):
//   phase 0: async-stage Q strip (32 KB) into LDS, shared by all 8 waves
//   phase 1: WMMA Q·K^T, 4 accumulators/wave (A-fragment reuse x4),
//            16x2048 f32 score strip lives in LDS (128 KB)
//   phase 2: iterative top-32 extraction per row (wave butterfly reduce)
//   phase 3: normalize by sum(top-k)+1e-10, gather-accumulate V rows
// ---------------------------------------------------------------------------
__global__ __launch_bounds__(256) void sparse_attn_scores(
    const _Float16* __restrict__ Qh, const _Float16* __restrict__ Kh,
    const _Float16* __restrict__ Vh, float* __restrict__ out)
{
    extern __shared__ char smem[];
    _Float16* qs = (_Float16*)smem;                       // [16][EE]  32 KB
    float* sc    = (float*)(smem + (size_t)16 * EE * 2);  // [16][SS] 128 KB
    float* topv  = sc + 16 * SS;                          // [16][TOPK]
    int*   topi  = (int*)(topv + 16 * TOPK);              // [16][TOPK]

    const int tid  = threadIdx.x;
    const int wave = tid >> 5;
    const int lane = tid & 31;
    const int lm   = lane & 15;
    const int hi   = lane >> 4;

    const int b  = blockIdx.y;
    const int q0 = blockIdx.x * 16;

    const _Float16* qg    = Qh + ((size_t)b * SS + q0) * EE;
    const _Float16* kbase = Kh + (size_t)b * SS * EE;

    // ---- phase 0: stage Q strip into LDS (async, tracked by ASYNCcnt) ----
#ifdef HAVE_ASYNC_LDS
    {
        gv4i_p gsrc = (gv4i_p)qg;
        lv4i_p ldst = (lv4i_p)qs;
        // 16*EE f16 = 32 KB = 2048 x 16-byte chunks; 256 threads x 8 chunks
        for (int i = tid; i < (16 * EE) / 8; i += 256)
            __builtin_amdgcn_global_load_async_to_lds_b128(gsrc + i, ldst + i, 0, 0);
        __builtin_amdgcn_s_wait_asynccnt(0);
    }
#else
    for (int i = tid * 8; i < 16 * EE; i += 256 * 8)
        *(v8h*)(qs + i) = *(const v8h*)(qg + i);
#endif
    __syncthreads();

    // ---- phase 1: scores (4 N-tiles per wave per pass) ----
    // A fragments come from LDS; lane holds M=lm, contiguous K run offset by 16*hi.
    const _Float16* aq = qs + (size_t)lm * EE + hi * 16;

    for (int pass = 0; pass < 4; ++pass) {
        const int jbase = pass * 512 + wave * 16;
        const _Float16* br0 = kbase + (size_t)(jbase +   0 + lm) * EE + hi * 16;
        const _Float16* br1 = kbase + (size_t)(jbase + 128 + lm) * EE + hi * 16;
        const _Float16* br2 = kbase + (size_t)(jbase + 256 + lm) * EE + hi * 16;
        const _Float16* br3 = kbase + (size_t)(jbase + 384 + lm) * EE + hi * 16;

        v8f c0 = {}, c1 = {}, c2 = {}, c3 = {};
        for (int k0 = 0; k0 < EE; k0 += 32) {
            v8h a0 = *(const v8h*)(aq + k0);        // ds_load_b128
            v8h a1 = *(const v8h*)(aq + k0 + 8);
            v16h a;
#pragma unroll
            for (int i = 0; i < 8; ++i) { a[i] = a0[i]; a[8 + i] = a1[i]; }
            v16h b0 = *(const v16h*)(br0 + k0);
            v16h b1 = *(const v16h*)(br1 + k0);
            v16h b2 = *(const v16h*)(br2 + k0);
            v16h b3 = *(const v16h*)(br3 + k0);
            __builtin_prefetch(br0 + k0 + 64, 0, 1);
            c0 = __builtin_amdgcn_wmma_f32_16x16x32_f16(false, a, false, b0,
                                                        (short)0, c0, false, false);
            c1 = __builtin_amdgcn_wmma_f32_16x16x32_f16(false, a, false, b1,
                                                        (short)0, c1, false, false);
            c2 = __builtin_amdgcn_wmma_f32_16x16x32_f16(false, a, false, b2,
                                                        (short)0, c2, false, false);
            c3 = __builtin_amdgcn_wmma_f32_16x16x32_f16(false, a, false, b3,
                                                        (short)0, c3, false, false);
        }
#pragma unroll
        for (int r = 0; r < 8; ++r) {
            const size_t rowoff = (size_t)(hi * 8 + r) * SS;
            sc[rowoff + jbase +   0 + lm] = c0[r];
            sc[rowoff + jbase + 128 + lm] = c1[r];
            sc[rowoff + jbase + 256 + lm] = c2[r];
            sc[rowoff + jbase + 384 + lm] = c3[r];
        }
    }
    __syncthreads();

    // ---- phase 2: top-32 per row; wave handles 2 rows ----
    for (int rr = 0; rr < 2; ++rr) {
        const int row = wave * 2 + rr;
        volatile float* srow = sc + (size_t)row * SS;
        for (int p = 0; p < TOPK; ++p) {
            float v = -3.402823466e38f;
            int   idx = 0;
            for (int t = lane; t < SS; t += 32) {
                float s = srow[t];
                if (s > v) { v = s; idx = t; }
            }
#pragma unroll
            for (int off = 16; off > 0; off >>= 1) {
                float ov = __shfl_xor(v, off, 32);
                int   oi = __shfl_xor(idx, off, 32);
                if (ov > v || (ov == v && oi < idx)) { v = ov; idx = oi; }
            }
            // all lanes converged to same (v, idx); lane 0 records + invalidates
            if (lane == 0) {
                topv[row * TOPK + p] = v;
                topi[row * TOPK + p] = idx;
                srow[idx] = -3.402823466e38f;   // same-wave LDS ops are in-order
            }
        }
        // normalize: denom = sum(top-k) + 1e-10 (reference does NOT exp)
        float tv  = topv[row * TOPK + lane];
        float sum = tv;
#pragma unroll
        for (int off = 16; off > 0; off >>= 1) sum += __shfl_xor(sum, off, 32);
        topv[row * TOPK + lane] = tv / (sum + 1e-10f);
    }
    __syncthreads();

    // ---- phase 3: out[q] = sum_j w_j * V[idx_j]  (coalesced over e) ----
    const _Float16* vbase = Vh + (size_t)b * SS * EE;
    float* obase = out + ((size_t)b * SS + q0) * EE;
    for (int oi = tid; oi < 16 * EE; oi += 256) {
        const int r = oi >> 10;          // EE == 1024
        const int e = oi & (EE - 1);
        float acc = 0.f;
#pragma unroll 8
        for (int j = 0; j < TOPK; ++j) {
            acc += topv[r * TOPK + j]
                 * (float)vbase[(size_t)topi[r * TOPK + j] * EE + e];
        }
        obase[(size_t)r * EE + e] = acc;
    }
}

// ---------------------------------------------------------------------------
extern "C" void kernel_launch(void* const* d_in, const int* in_sizes, int n_in,
                              void* d_out, int out_size, void* d_ws, size_t ws_size,
                              hipStream_t stream) {
    (void)in_sizes; (void)n_in; (void)out_size; (void)ws_size;
    const float* x  = (const float*)d_in[0];
    const float* Wq = (const float*)d_in[1];
    const float* bq = (const float*)d_in[2];
    const float* Wk = (const float*)d_in[3];
    const float* bk = (const float*)d_in[4];
    const float* Wv = (const float*)d_in[5];
    const float* bv = (const float*)d_in[6];
    float* out = (float*)d_out;

    char* ws = (char*)d_ws;
    const size_t bytesH = (size_t)MTOT * EE * sizeof(_Float16);   // 16 MB each
    _Float16* Qh = (_Float16*)(ws);
    _Float16* Kh = (_Float16*)(ws + bytesH);
    _Float16* Vh = (_Float16*)(ws + 2 * bytesH);                  // 48 MB total

    sparse_attn_qkv<<<dim3(MTOT / 16, EE / 128, 3), 256, 0, stream>>>(
        x, Wq, bq, Wk, bk, Wv, bv, Qh, Kh, Vh);

    const size_t smem = (size_t)16 * EE * sizeof(_Float16)        // Q strip
                      + (size_t)16 * SS * sizeof(float)           // score strip
                      + (size_t)16 * TOPK * (sizeof(float) + sizeof(int));
    sparse_attn_scores<<<dim3(SS / 16, BB), 256, smem, stream>>>(Qh, Kh, Vh, out);
}
